// QwenImageCrossAttention_76081050681773
// MI455X (gfx1250) — matrix operand order repaired and checked by reading, not compiled, not run on hardware
//
#include <hip/hip_runtime.h>

#define DIM   3072
#define NH    24
#define DH    128
#define QS    (NH * DH)            // 3072
#define S_IMG 2048
#define S_TXT 256
#define S_ALL (S_IMG + S_TXT)      // 2304
#define EPSF  1e-6f
#define SCALEF 0.08838834764831845f  // 1/sqrt(128)

typedef __attribute__((ext_vector_type(16))) __bf16 bf16x16;
typedef __attribute__((ext_vector_type(8)))  __bf16 bf16x8;
typedef __attribute__((ext_vector_type(4)))  __bf16 bf16x4;
typedef __attribute__((ext_vector_type(2)))  __bf16 bf16x2;
typedef __attribute__((ext_vector_type(8)))  float  f32x8;
typedef __attribute__((ext_vector_type(4)))  float  f32x4;

__device__ __forceinline__ f32x8 wmma_bf16(bf16x16 a, bf16x16 b, f32x8 c) {
    // D = A(16x32 bf16) * B(32x16 bf16) + C(16x16 f32)
    return __builtin_amdgcn_wmma_f32_16x16x32_bf16(
        false, a, false, b, (short)0, c, false, false);
}

__device__ __forceinline__ bf16x16 cat8(bf16x8 lo, bf16x8 hi) {
    return __builtin_shufflevector(lo, hi, 0, 1, 2, 3, 4, 5, 6, 7,
                                           8, 9, 10, 11, 12, 13, 14, 15);
}

// A-fragment (16x32 bf16) from a row-major tile (lda elements per row).
// ISA 7.12.2: lane holds row M=lane&15; lo half needs K runs [0..7],[16..23],
// hi half [8..15],[24..31] -> two contiguous 8-element (16B) loads per lane.
__device__ __forceinline__ bf16x16 load_a_frag(const __bf16* tile, int lda) {
    const int lane = threadIdx.x & 31;
    const __bf16* p = tile + (lane & 15) * lda + (lane >> 4) * 8;
    bf16x8 v0 = *(const bf16x8*)(p);
    bf16x8 v1 = *(const bf16x8*)(p + 16);
    return cat8(v0, v1);
}

// B-fragment (32x16 bf16) from a K-contiguous-per-column tile: tileT[n][k],
// ldb elements per column-row. Lane holds col N=(lane&15)+ncol; lo half K runs
// [0..7],[16..23], hi half [8..15],[24..31] -> two 16B loads per lane.
__device__ __forceinline__ bf16x16 load_bT_frag(const __bf16* tileT, int ldb, int ncol) {
    const int lane = threadIdx.x & 31;
    const __bf16* p = tileT + ((lane & 15) + ncol) * ldb + (lane >> 4) * 8;
    bf16x8 v0 = *(const bf16x8*)(p);
    bf16x8 v1 = *(const bf16x8*)(p + 16);
    return cat8(v0, v1);
}

// ---------------------------------------------------------------------------
// GEMM: C[M,N] = A[M,K]*B[K,N] + bias[N]; fp32 in HBM, bf16 WMMA via LDS.
// Block tile 128x64 (8 waves, wave tile 32x32 = 4 WMMA / K-step), K-step 32.
// As row-major [128][32]; Bs transposed [64][32] so both frag loads are b128.
// Requires M%128==0, N%64==0, K%32==0 (true at all call sites).
// ---------------------------------------------------------------------------
__global__ __launch_bounds__(256)
void gemm_bias_kernel(const float* __restrict__ A, const float* __restrict__ B,
                      const float* __restrict__ bias, float* __restrict__ C,
                      int M, int N, int K) {
    __shared__ __attribute__((aligned(16))) __bf16 As[128 * 32];
    __shared__ __attribute__((aligned(16))) __bf16 BsT[64 * 32];
    const int tid  = threadIdx.x;
    const int lane = tid & 31;
    const int wave = tid >> 5;
    const int bm = blockIdx.y * 128;
    const int bn = blockIdx.x * 64;
    const int wm = (wave & 3) * 32;
    const int wn = (wave >> 2) * 32;

    // fill indices (hoisted)
    const int ar  = tid >> 1;            // 0..127 : A row
    const int acb = (tid & 1) * 16;      // 0/16   : A col base (16 floats)
    const int bkp = tid & 15;            // k-pair for B transpose fill
    const int bnb = tid >> 4;            // 0..15? no: 256 threads -> 0..15
    // B fill: thread t handles k pair (2*bkp, 2*bkp+1), n slice bnb*4..+3
    // (16 kp x 16 nb covers 32k x 64n)

    f32x8 c00 = {}, c01 = {}, c10 = {}, c11 = {};

    for (int k0 = 0; k0 < K; k0 += 32) {
        if (k0 + 32 < K) {   // prefetch next K-slab (global_prefetch_b8)
            __builtin_prefetch(A + (size_t)(bm + ar) * K + k0 + 32 + acb, 0, 1);
            __builtin_prefetch(B + (size_t)(k0 + 32 + 2 * bkp) * N + bn + bnb * 4, 0, 1);
        }
        __syncthreads();
        {   // A tile 128x32: 16 floats -> two 16B bf16 stores per thread
            const float* src = A + (size_t)(bm + ar) * K + k0 + acb;
            f32x4 f0 = *(const f32x4*)(src + 0);
            f32x4 f1 = *(const f32x4*)(src + 4);
            f32x4 f2 = *(const f32x4*)(src + 8);
            f32x4 f3 = *(const f32x4*)(src + 12);
            bf16x8 h0, h1;
#pragma unroll
            for (int e = 0; e < 4; ++e) {
                h0[e]     = (__bf16)f0[e];
                h0[e + 4] = (__bf16)f1[e];
                h1[e]     = (__bf16)f2[e];
                h1[e + 4] = (__bf16)f3[e];
            }
            *(bf16x8*)(As + ar * 32 + acb)     = h0;
            *(bf16x8*)(As + ar * 32 + acb + 8) = h1;
        }
        {   // B tile 32x64 stored transposed: pack (k,k+1) pairs -> b32 stores
            const float* r0 = B + (size_t)(k0 + 2 * bkp) * N + bn + bnb * 4;
            const float* r1 = r0 + N;
            f32x4 f0 = *(const f32x4*)r0;
            f32x4 f1 = *(const f32x4*)r1;
#pragma unroll
            for (int e = 0; e < 4; ++e) {
                bf16x2 pk = { (__bf16)f0[e], (__bf16)f1[e] };
                *(bf16x2*)(BsT + (bnb * 4 + e) * 32 + 2 * bkp) = pk;
            }
        }
        __syncthreads();

        bf16x16 a0 = load_a_frag(As + (wm + 0)  * 32, 32);
        bf16x16 a1 = load_a_frag(As + (wm + 16) * 32, 32);
        bf16x16 b0 = load_bT_frag(BsT, 32, wn);
        bf16x16 b1 = load_bT_frag(BsT, 32, wn + 16);
        c00 = wmma_bf16(a0, b0, c00);
        c01 = wmma_bf16(a0, b1, c01);
        c10 = wmma_bf16(a1, b0, c10);
        c11 = wmma_bf16(a1, b1, c11);
    }

    const int hi  = lane >> 4;
    const int col = lane & 15;
    const int n0 = bn + wn + col;
    const int n1 = n0 + 16;
    const float bia0 = bias[n0];
    const float bia1 = bias[n1];
#pragma unroll
    for (int r = 0; r < 8; ++r) {
        int m0 = bm + wm + r + hi * 8;     // C row layout: M = vgpr + 8*hi
        int m1 = m0 + 16;
        C[(size_t)m0 * N + n0] = c00[r] + bia0;
        C[(size_t)m0 * N + n1] = c01[r] + bia1;
        C[(size_t)m1 * N + n0] = c10[r] + bia0;
        C[(size_t)m1 * N + n1] = c11[r] + bia1;
    }
}

// ---------------------------------------------------------------------------
// Per-head RMSNorm + non-NeoX RoPE, fp32 qkv -> bf16 Q/K/V [tok][head][dh].
// One wave per (token, head); vectorized 16B loads / 8B stores.
// ---------------------------------------------------------------------------
__global__ __launch_bounds__(128)
void qkv_post_kernel(const float* __restrict__ qkv, int tokOff,
                     const float* __restrict__ cosT, const float* __restrict__ sinT,
                     const float* __restrict__ nqw, const float* __restrict__ nkw,
                     __bf16* __restrict__ qb, __bf16* __restrict__ kb,
                     __bf16* __restrict__ vb) {
    const int lane = threadIdx.x;                  // 0..31
    const int head = blockIdx.y * 4 + threadIdx.y; // 0..23
    const int tok  = blockIdx.x;
    const float* base = qkv + (size_t)tok * (3 * QS);
    const float c0 = cosT[tok * 64 + lane * 2 + 0];
    const float c1 = cosT[tok * 64 + lane * 2 + 1];
    const float s0 = sinT[tok * 64 + lane * 2 + 0];
    const float s1 = sinT[tok * 64 + lane * 2 + 1];
    const f32x4 nwq = *(const f32x4*)(nqw + lane * 4);
    const f32x4 nwk = *(const f32x4*)(nkw + lane * 4);
    const size_t outBase = ((size_t)(tokOff + tok) * NH + head) * DH + lane * 4;

#pragma unroll
    for (int which = 0; which < 2; ++which) {      // 0 = Q, 1 = K
        f32x4 x = *(const f32x4*)(base + which * QS + head * DH + lane * 4);
        float ss = x[0] * x[0] + x[1] * x[1] + x[2] * x[2] + x[3] * x[3];
#pragma unroll
        for (int m = 1; m < 32; m <<= 1) ss += __shfl_xor(ss, m, 32);
        float inv = rsqrtf(ss * (1.0f / DH) + EPSF);
        f32x4 nw = which ? nwk : nwq;
        float x0 = x[0] * inv * nw[0];
        float x1 = x[1] * inv * nw[1];
        float x2 = x[2] * inv * nw[2];
        float x3 = x[3] * inv * nw[3];
        bf16x4 o;
        o[0] = (__bf16)(x0 * c0 - x1 * s0);
        o[1] = (__bf16)(x1 * c0 + x0 * s0);
        o[2] = (__bf16)(x2 * c1 - x3 * s1);
        o[3] = (__bf16)(x3 * c1 + x2 * s1);
        *(bf16x4*)((which ? kb : qb) + outBase) = o;
    }
    f32x4 v = *(const f32x4*)(base + 2 * QS + head * DH + lane * 4);
    bf16x4 vo;
#pragma unroll
    for (int i = 0; i < 4; ++i) vo[i] = (__bf16)v[i];
    *(bf16x4*)(vb + outBase) = vo;
}

// ---------------------------------------------------------------------------
// Flash attention: 4 waves/block share one head's 32-key K/V tiles in LDS;
// each wave owns a 16-query tile. K staged row-major [key][dh] (b128 frag
// loads for QK^T); V staged transposed [dh][key] (b128 frag loads for P*V).
// 16 WMMA per wave per 32-key tile; online softmax in the C-layout rows.
// ---------------------------------------------------------------------------
__global__ __launch_bounds__(128)
void flash_attn_kernel(const __bf16* __restrict__ Q, const __bf16* __restrict__ Kb,
                       const __bf16* __restrict__ Vb, float* __restrict__ O) {
    __shared__ __attribute__((aligned(16))) __bf16 Ks [32 * 128];   // [key][dh]
    __shared__ __attribute__((aligned(16))) __bf16 VsT[128 * 32];   // [dh][key]
    __shared__ __attribute__((aligned(16))) __bf16 Plds[4 * 16 * 32];
    const int tid  = threadIdx.x;
    const int lane = tid & 31;
    const int wave = tid >> 5;
    const int hi   = lane >> 4;
    const int col  = lane & 15;
    const int head = blockIdx.y;
    const int q0   = (blockIdx.x * 4 + wave) * 16;
    const size_t hoff = (size_t)head * DH;
    __bf16* Pw = Plds + wave * (16 * 32);

    // K-tile fill indices: thread -> (key row, 32-dh slice)
    const int fkey  = tid >> 2;          // 0..31
    const int fpart = tid & 3;           // 0..3
    // V-tile (transposed) fill indices: thread -> (key pair, 16-dh slice)
    const int vkp   = tid & 15;          // keys 2*vkp, 2*vkp+1
    const int vpart = tid >> 4;          // 0..7 -> dh slice of 16

    // Q fragments: 4 K-chunks of 32, loaded as 2 x b128 per chunk per lane
    bf16x16 aq[4];
#pragma unroll
    for (int ch = 0; ch < 4; ++ch)
        aq[ch] = load_a_frag(Q + (size_t)q0 * QS + hoff + ch * 32, QS);

    f32x8 o[8];
    const f32x8 zf = {};
#pragma unroll
    for (int i = 0; i < 8; ++i) o[i] = zf;
    float mrow[8], lrow[8];
#pragma unroll
    for (int r = 0; r < 8; ++r) { mrow[r] = -1e30f; lrow[r] = 0.0f; }

    for (int kt = 0; kt < S_ALL / 32; ++kt) {
        const int ktb = kt * 32;
        __syncthreads();
        {   // stage K tile row-major: 4 x (b128 load + b128 store)
            const __bf16* src = Kb + (size_t)(ktb + fkey) * QS + hoff + fpart * 32;
            __bf16* dst = Ks + fkey * 128 + fpart * 32;
#pragma unroll
            for (int e = 0; e < 4; ++e)
                *(bf16x8*)(dst + e * 8) = *(const bf16x8*)(src + e * 8);
        }
        {   // stage V tile transposed: pack key pairs -> b32 stores
            const __bf16* r0 = Vb + (size_t)(ktb + 2 * vkp) * QS + hoff + vpart * 16;
            const __bf16* r1 = r0 + QS;
            bf16x8 a0 = *(const bf16x8*)(r0);
            bf16x8 a1 = *(const bf16x8*)(r0 + 8);
            bf16x8 b0 = *(const bf16x8*)(r1);
            bf16x8 b1 = *(const bf16x8*)(r1 + 8);
#pragma unroll
            for (int e = 0; e < 8; ++e) {
                bf16x2 p0 = { a0[e], b0[e] };
                bf16x2 p1 = { a1[e], b1[e] };
                *(bf16x2*)(VsT + (vpart * 16 + e)     * 32 + 2 * vkp) = p0;
                *(bf16x2*)(VsT + (vpart * 16 + e + 8) * 32 + 2 * vkp) = p1;
            }
        }
        __syncthreads();

        // QK^T: S tile 16x32 = two accumulators, 8 WMMA
        f32x8 s0 = {}, s1 = {};
#pragma unroll
        for (int ch = 0; ch < 4; ++ch) {
            bf16x16 b0 = load_bT_frag(Ks + ch * 32, 128, 0);
            bf16x16 b1 = load_bT_frag(Ks + ch * 32, 128, 16);
            s0 = wmma_bf16(aq[ch], b0, s0);
            s1 = wmma_bf16(aq[ch], b1, s1);
        }
        // online softmax; element r <-> score row (r + 8*hi)
#pragma unroll
        for (int r = 0; r < 8; ++r) {
            float v0 = s0[r] * SCALEF;
            float v1 = s1[r] * SCALEF;
            float mx = fmaxf(v0, v1);
#pragma unroll
            for (int m = 1; m < 16; m <<= 1) mx = fmaxf(mx, __shfl_xor(mx, m, 32));
            float mnew  = fmaxf(mrow[r], mx);
            float alpha = __expf(mrow[r] - mnew);
            float p0 = __expf(v0 - mnew);
            float p1 = __expf(v1 - mnew);
            float rs = p0 + p1;
#pragma unroll
            for (int m = 1; m < 16; m <<= 1) rs += __shfl_xor(rs, m, 32);
            lrow[r] = lrow[r] * alpha + rs;
            mrow[r] = mnew;
#pragma unroll
            for (int nc = 0; nc < 8; ++nc) o[nc][r] *= alpha;
            int prow = r + 8 * hi;
            Pw[prow * 32 + col]      = (__bf16)p0;
            Pw[prow * 32 + 16 + col] = (__bf16)p1;
        }
        // P*V: LDS ops are in-order within a wave; Pw is wave-private
        bf16x16 pa = load_a_frag(Pw, 32);
#pragma unroll
        for (int nc = 0; nc < 8; ++nc) {
            bf16x16 bv = load_bT_frag(VsT, 32, nc * 16);
            o[nc] = wmma_bf16(pa, bv, o[nc]);
        }
    }

#pragma unroll
    for (int r = 0; r < 8; ++r) {
        float rl = 1.0f / lrow[r];
        int tok = q0 + r + 8 * hi;
#pragma unroll
        for (int nc = 0; nc < 8; ++nc)
            O[(size_t)tok * QS + hoff + nc * 16 + col] = o[nc][r] * rl;
    }
}

// ---------------------------------------------------------------------------
extern "C" void kernel_launch(void* const* d_in, const int* in_sizes, int n_in,
                              void* d_out, int out_size, void* d_ws, size_t ws_size,
                              hipStream_t stream) {
    const float* hs     = (const float*)d_in[0];
    const float* ehs    = (const float*)d_in[1];
    const float* vcos   = (const float*)d_in[2];
    const float* vsin   = (const float*)d_in[3];
    const float* tcos   = (const float*)d_in[4];
    const float* tsin   = (const float*)d_in[5];
    const float* w_qkv  = (const float*)d_in[6];
    const float* b_qkv  = (const float*)d_in[7];
    const float* w_aqkv = (const float*)d_in[8];
    const float* b_aqkv = (const float*)d_in[9];
    const float* nqw    = (const float*)d_in[10];
    const float* nkw    = (const float*)d_in[11];
    const float* naqw   = (const float*)d_in[12];
    const float* nakw   = (const float*)d_in[13];
    const float* w_out  = (const float*)d_in[14];
    const float* b_out  = (const float*)d_in[15];
    const float* w_aout = (const float*)d_in[16];
    const float* b_aout = (const float*)d_in[17];

    char* ws = (char*)d_ws;
    float*  img_qkv = (float*)ws;  ws += (size_t)S_IMG * 3 * QS * sizeof(float);
    float*  txt_qkv = (float*)ws;  ws += (size_t)S_TXT * 3 * QS * sizeof(float);
    __bf16* qb      = (__bf16*)ws; ws += (size_t)S_ALL * QS * sizeof(__bf16);
    __bf16* kb      = (__bf16*)ws; ws += (size_t)S_ALL * QS * sizeof(__bf16);
    __bf16* vb      = (__bf16*)ws; ws += (size_t)S_ALL * QS * sizeof(__bf16);
    float*  attn    = (float*)ws;  ws += (size_t)S_ALL * QS * sizeof(float);

    float* img_out = (float*)d_out;
    float* txt_out = img_out + (size_t)S_IMG * DIM;

    // 1) QKV projections (bf16 WMMA, f32 accumulate, fused bias)
    gemm_bias_kernel<<<dim3(3 * QS / 64, S_IMG / 128), 256, 0, stream>>>(
        hs, w_qkv, b_qkv, img_qkv, S_IMG, 3 * QS, DIM);
    gemm_bias_kernel<<<dim3(3 * QS / 64, S_TXT / 128), 256, 0, stream>>>(
        ehs, w_aqkv, b_aqkv, txt_qkv, S_TXT, 3 * QS, DIM);

    // 2) RMSNorm + RoPE -> concatenated bf16 Q/K/V (txt tokens first)
    qkv_post_kernel<<<dim3(S_TXT, NH / 4), dim3(32, 4), 0, stream>>>(
        txt_qkv, 0, tcos, tsin, naqw, nakw, qb, kb, vb);
    qkv_post_kernel<<<dim3(S_IMG, NH / 4), dim3(32, 4), 0, stream>>>(
        img_qkv, S_TXT, vcos, vsin, nqw, nkw, qb, kb, vb);

    // 3) Flash attention: 4 query tiles per block share K/V LDS tiles
    flash_attn_kernel<<<dim3(S_ALL / 64, NH), 128, 0, stream>>>(qb, kb, vb, attn);

    // 4) Output projections (img rows start after the 256 txt rows)
    gemm_bias_kernel<<<dim3(DIM / 64, S_IMG / 128), 256, 0, stream>>>(
        attn + (size_t)S_TXT * QS, w_out, b_out, img_out, S_IMG, DIM, QS);
    gemm_bias_kernel<<<dim3(DIM / 64, S_TXT / 128), 256, 0, stream>>>(
        attn, w_aout, b_aout, txt_out, S_TXT, DIM, QS);
}
